// RNN_wrapper_90589450207796
// MI455X (gfx1250) — matrix operand order repaired and checked by reading, not compiled, not run on hardware
//
#include <hip/hip_runtime.h>
#include <hip/hip_bf16.h>

// ---------------------------------------------------------------------------
// 2-layer GRU (T=512, B=64, D=512, H=1024) for gfx1250 (MI455X).
//
//  * f16 weights (converted once, ~22MB, L2-resident on 192MB L2); row-major
//    W rows are WMMA B-columns of W^T -> coalesced 32B loads, no transpose.
//  * Per step/layer: fused GEMM [x_t | h_prev] @ [W_ih | W_hh]^T with
//    v_wmma_f32_16x16x32_f16 (f32 accum). n-gate keeps x-part and h-part
//    separate (n = tanh(xn + r*hn)).
//  * 8 waves/WG: unified K space split across two wave-groups (halves the
//    serial WMMA chain, doubles latency-hiding waves); partials reduced
//    through LDS. Wave id is readfirstlane'd -> all loop bounds are SGPRs,
//    scalar loop control, EXEC all-1s for every WMMA.
//  * Software-pipelined fragment loads with running per-lane pointers
//    (pointer += 32 per chunk, no in-loop address rebuild), unroll 2.
//  * Layers pipelined per time step -> workspace ~24MB, 1024 graph-friendly
//    launches on `stream`.
// ---------------------------------------------------------------------------

#define TT 512
#define BB 64
#define DD 512
#define HH 1024

typedef __attribute__((ext_vector_type(16))) _Float16 v16h;
typedef __attribute__((ext_vector_type(8)))  float    v8f;

__device__ __forceinline__ float sigmoid_f(float x) {
  return 1.0f / (1.0f + __expf(-x));
}
__device__ __forceinline__ float tanh_f(float x) {
  return 2.0f / (1.0f + __expf(-2.0f * x)) - 1.0f;
}

__device__ __forceinline__ v8f wmma_f16(v16h a, v16h b, v8f c) {
  return __builtin_amdgcn_wmma_f32_16x16x32_f16(
      /*neg_a=*/false, a, /*neg_b=*/false, b,
      /*c_mod=*/(short)0, c, /*reuse_a=*/false, /*reuse_b=*/false);
}

// A-fragment (16x32 f16) from an f32 row-major matrix, p already points at
// row(M=lane%16) + k0 + 16*(lane>=16). Elems 0-15 at p[0..15].
__device__ __forceinline__ v16h load_a_ptr(const float* __restrict__ p) {
  const float4 x0 = ((const float4*)p)[0];
  const float4 x1 = ((const float4*)p)[1];
  const float4 y0 = ((const float4*)(p + 8))[0];
  const float4 y1 = ((const float4*)(p + 8))[1];
  v16h a;
  a[0]  = (_Float16)x0.x; a[1]  = (_Float16)x0.y;
  a[2]  = (_Float16)x0.z; a[3]  = (_Float16)x0.w;
  a[4]  = (_Float16)x1.x; a[5]  = (_Float16)x1.y;
  a[6]  = (_Float16)x1.z; a[7]  = (_Float16)x1.w;
  a[8]  = (_Float16)y0.x; a[9]  = (_Float16)y0.y;
  a[10] = (_Float16)y0.z; a[11] = (_Float16)y0.w;
  a[12] = (_Float16)y1.x; a[13] = (_Float16)y1.y;
  a[14] = (_Float16)y1.z; a[15] = (_Float16)y1.w;
  return a;
}

// B-fragment (32x16 f16): p points at W row(N=lane%16) + k0 + 16*(lane>=16);
// one contiguous 32B load.
__device__ __forceinline__ v16h load_b_ptr(const _Float16* __restrict__ p) {
  return *(const v16h*)p;
}

// Pipelined GEMM over K range [kBeg, kEnd) (multiples of 32; SGPR bounds),
// accumulating three gate columns (r, z, n-part) for one 16x16 tile.
// Running per-lane pointers advance by 32 elements per chunk.
__device__ __forceinline__ void gemm_range(const float* __restrict__ A,
                                           long long aStride,
                                           const _Float16* __restrict__ W,
                                           int K,  // row length of W
                                           int kBeg, int kEnd,
                                           int m0, int j0, int lane,
                                           v8f& accR, v8f& accZ, v8f& accN) {
  if (kBeg >= kEnd) return;
  const int mrow = m0 + (lane & 15);
  const int ncol = j0 + (lane & 15);
  const int kh8  = (lane >> 4) << 3;   // unused (0 or 8); A and B both use kh16
  const int kh16 = (lane >> 4) << 4;   // B half-select: 0 or 16
  const float*    ap = A + (long long)mrow * aStride + kBeg + kh16;
  const _Float16* wr = W + ((long long)0 * HH + ncol) * K + kBeg + kh16;
  const _Float16* wz = W + ((long long)1 * HH + ncol) * K + kBeg + kh16;
  const _Float16* wn = W + ((long long)2 * HH + ncol) * K + kBeg + kh16;

  v16h a0  = load_a_ptr(ap);
  v16h br0 = load_b_ptr(wr);
  v16h bz0 = load_b_ptr(wz);
  v16h bn0 = load_b_ptr(wn);
#pragma unroll 2
  for (int k = kBeg + 32; k < kEnd; k += 32) {
    ap += 32; wr += 32; wz += 32; wn += 32;
    // Issue next chunk's loads before consuming the current chunk.
    v16h a1  = load_a_ptr(ap);
    v16h br1 = load_b_ptr(wr);
    v16h bz1 = load_b_ptr(wz);
    v16h bn1 = load_b_ptr(wn);
    accR = wmma_f16(a0, br0, accR);
    accZ = wmma_f16(a0, bz0, accZ);
    accN = wmma_f16(a0, bn0, accN);
    a0 = a1; br0 = br1; bz0 = bz1; bn0 = bn1;
  }
  accR = wmma_f16(a0, br0, accR);
  accZ = wmma_f16(a0, bz0, accZ);
  accN = wmma_f16(a0, bn0, accN);
}

// One GRU time step for one layer (PyTorch gate order r,z,n):
//   r = sigm(xr+hr), z = sigm(xz+hz), n = tanh(xn + r*hn),
//   h = (1-z)*n + z*h_prev
// Grid: HH/16 blocks x 256 threads (8 waves).
//   wave&3  -> M-tile (batch rows 16*(wave&3) ..+15)
//   wave>>2 -> K-half of the unified [input | hidden] K space (LDS-reduced)
template <int SEG1_LEN>
__global__ __launch_bounds__(256)
void gru_step_kernel(const float* __restrict__ seg1,      // x_t rows (f32)
                     long long seg1_stride,
                     const float* __restrict__ hprev,     // (B,H) f32
                     const _Float16* __restrict__ Wih,    // (3H, SEG1_LEN) f16
                     const _Float16* __restrict__ Whh,    // (3H, H) f16
                     const float* __restrict__ bih,
                     const float* __restrict__ bhh,
                     float* __restrict__ hout,            // (B,H) f32
                     float* __restrict__ final_out,       // (B,H) or null
                     const int* __restrict__ length,      // (B,) or null
                     int t) {
  const int lane = threadIdx.x & 31;
  // Wave id is wave-uniform: force it into an SGPR so all derived loop
  // bounds are scalar (scalar loop control; EXEC stays all-1s for WMMA).
  const int wave   = __builtin_amdgcn_readfirstlane((int)(threadIdx.x >> 5));
  const int mwave  = wave & 3;
  const int kgroup = wave >> 2;
  const int m0 = mwave * 16;
  const int j0 = blockIdx.x * 16;

  // Unified K space: [0, SEG1_LEN) = input segment, [SEG1_LEN, kTotal) =
  // hidden segment. Split in half across the two wave groups; clamp each
  // half against the segment boundary (constant-folded scalar math).
  constexpr int kTotal = SEG1_LEN + HH;  // half is a multiple of 32
  constexpr int half = kTotal / 2;
  const int kBeg = kgroup * half;
  const int kEnd = kBeg + half;
  const int s1Beg = (kBeg < SEG1_LEN) ? kBeg : SEG1_LEN;
  const int s1End = (kEnd < SEG1_LEN) ? kEnd : SEG1_LEN;
  const int s2Beg = ((kBeg > SEG1_LEN) ? kBeg : SEG1_LEN) - SEG1_LEN;
  const int s2End = ((kEnd > SEG1_LEN) ? kEnd : SEG1_LEN) - SEG1_LEN;

  v8f accR = {}, accZ = {}, accXN = {}, accHN = {};

  // Input segment: accumulate r, z, and the x-part of n.
  gemm_range(seg1, seg1_stride, Wih, SEG1_LEN, s1Beg, s1End,
             m0, j0, lane, accR, accZ, accXN);
  // Hidden segment: accumulate r, z, and the h-part of n.
  gemm_range(hprev, (long long)HH, Whh, HH, s2Beg, s2End,
             m0, j0, lane, accR, accZ, accHN);

  // Cross-wave-group reduction through LDS (16KB).
  __shared__ v8f red[4][4][32];
  if (kgroup == 1) {
    red[mwave][0][lane] = accR;
    red[mwave][1][lane] = accZ;
    red[mwave][2][lane] = accXN;
    red[mwave][3][lane] = accHN;
  }
  __syncthreads();
  if (kgroup == 0) {
    accR  = accR  + red[mwave][0][lane];
    accZ  = accZ  + red[mwave][1][lane];
    accXN = accXN + red[mwave][2][lane];
    accHN = accHN + red[mwave][3][lane];

    // Epilogue. C/D layout: VGPR r -> M = r + 8*(lane>=16), N = lane%16.
    const int jn = j0 + (lane & 15);
    const int bbase = m0 + ((lane >> 4) << 3);
    const float br_ = bih[jn] + bhh[jn];
    const float bz_ = bih[HH + jn] + bhh[HH + jn];
    const float bxn = bih[2 * HH + jn];
    const float bhn = bhh[2 * HH + jn];
#pragma unroll
    for (int r = 0; r < 8; ++r) {
      const int b = bbase + r;
      const float rg = sigmoid_f(accR[r] + br_);
      const float zg = sigmoid_f(accZ[r] + bz_);
      const float ng = tanh_f(accXN[r] + bxn + rg * (accHN[r] + bhn));
      const float hp = hprev[(long long)b * HH + jn];
      const float hn = (1.0f - zg) * ng + zg * hp;
      hout[(long long)b * HH + jn] = hn;
      if (final_out != nullptr) {
        if (t == length[b] - 1) final_out[b * HH + jn] = hn;
      }
    }
  }
}

__global__ void cvt_f32_to_f16_kernel(const float* __restrict__ src,
                                      _Float16* __restrict__ dst, int n) {
  const int i = blockIdx.x * blockDim.x + threadIdx.x;
  if (i < n) dst[i] = (_Float16)src[i];
}

__global__ void zero_f32_kernel(float* __restrict__ p, int n) {
  const int i = blockIdx.x * blockDim.x + threadIdx.x;
  if (i < n) p[i] = 0.0f;
}

extern "C" void kernel_launch(void* const* d_in, const int* in_sizes, int n_in,
                              void* d_out, int out_size, void* d_ws, size_t ws_size,
                              hipStream_t stream) {
  (void)in_sizes; (void)n_in; (void)out_size; (void)ws_size;

  const float* input  = (const float*)d_in[0];  // (B, T, D)
  const int*   length = (const int*)  d_in[1];  // (B,)
  const float* W_ih0  = (const float*)d_in[2];  // (3H, D)
  const float* W_hh0  = (const float*)d_in[3];  // (3H, H)
  const float* b_ih0  = (const float*)d_in[4];
  const float* b_hh0  = (const float*)d_in[5];
  const float* W_ih1  = (const float*)d_in[6];  // (3H, H)
  const float* W_hh1  = (const float*)d_in[7];  // (3H, H)
  const float* b_ih1  = (const float*)d_in[8];
  const float* b_hh1  = (const float*)d_in[9];
  float* out = (float*)d_out;                   // (B, H)

  // Workspace carve (~24 MB total)
  char* ws = (char*)d_ws;
  size_t off = 0;
  auto carve = [&](size_t bytes) -> void* {
    void* p = ws + off;
    off += (bytes + 255) & ~(size_t)255;
    return p;
  };
  _Float16* Wih0h = (_Float16*)carve((size_t)3 * HH * DD * sizeof(_Float16));
  _Float16* Whh0h = (_Float16*)carve((size_t)3 * HH * HH * sizeof(_Float16));
  _Float16* Wih1h = (_Float16*)carve((size_t)3 * HH * HH * sizeof(_Float16));
  _Float16* Whh1h = (_Float16*)carve((size_t)3 * HH * HH * sizeof(_Float16));
  float* h0a = (float*)carve((size_t)BB * HH * sizeof(float));
  float* h0b = (float*)carve((size_t)BB * HH * sizeof(float));
  float* h1a = (float*)carve((size_t)BB * HH * sizeof(float));
  float* h1b = (float*)carve((size_t)BB * HH * sizeof(float));

  // Weight conversion to f16 (re-done every call; deterministic).
  {
    int n = 3 * HH * DD;
    cvt_f32_to_f16_kernel<<<(n + 255) / 256, 256, 0, stream>>>(W_ih0, Wih0h, n);
    n = 3 * HH * HH;
    cvt_f32_to_f16_kernel<<<(n + 255) / 256, 256, 0, stream>>>(W_hh0, Whh0h, n);
    cvt_f32_to_f16_kernel<<<(n + 255) / 256, 256, 0, stream>>>(W_ih1, Wih1h, n);
    cvt_f32_to_f16_kernel<<<(n + 255) / 256, 256, 0, stream>>>(W_hh1, Whh1h, n);
  }
  // h0 = h1 = 0
  {
    const int n = BB * HH;
    zero_f32_kernel<<<(n + 255) / 256, 256, 0, stream>>>(h0a, n);
    zero_f32_kernel<<<(n + 255) / 256, 256, 0, stream>>>(h1a, n);
  }

  // Pipelined 2-layer scan: layer1(t) consumes layer0(t)'s output directly.
  float* h0p = h0a; float* h0c = h0b;
  float* h1p = h1a; float* h1c = h1b;
  const dim3 grid(HH / 16);
  const dim3 block(256);
  for (int t = 0; t < TT; ++t) {
    // Layer 0: seg1 = x_t, rows strided by T*D in the (B,T,D) input
    gru_step_kernel<DD><<<grid, block, 0, stream>>>(
        input + (long long)t * DD, (long long)TT * DD,
        h0p, Wih0h, Whh0h, b_ih0, b_hh0,
        h0c, nullptr, nullptr, t);
    // Layer 1: seg1 = layer0's output at time t (h0c), rows strided by H
    gru_step_kernel<HH><<<grid, block, 0, stream>>>(
        h0c, (long long)HH,
        h1p, Wih1h, Whh1h, b_ih1, b_hh1,
        h1c, out, length, t);
    float* tmp;
    tmp = h0p; h0p = h0c; h0c = tmp;
    tmp = h1p; h1p = h1c; h1c = tmp;
  }
}
